// Net_10462540333328
// MI455X (gfx1250) — matrix operand order repaired and checked by reading, not compiled, not run on hardware
//
#include <hip/hip_runtime.h>
#include <hip/hip_bf16.h>

// ---------------------------------------------------------------------------
// Problem constants (match reference)
// ---------------------------------------------------------------------------
#define NN   50000
#define EE   100000
#define CC   10000
#define KK   6
#define RR   (CC * KK)      // 60000
#define GG   1024
#define HH   128
#define H2   256
#define H3   384
#define BN_EPS 1e-5f

typedef __attribute__((ext_vector_type(16))) __bf16          v16bf;
typedef __attribute__((ext_vector_type(16))) unsigned short  v16u;
typedef __attribute__((ext_vector_type(8)))  unsigned short  v8u;
typedef __attribute__((ext_vector_type(8)))  float           v8f;

// f32 -> bf16 round-to-nearest-even (manual; proven to compile)
__device__ __forceinline__ unsigned short f2bf(float f) {
  unsigned int u = __builtin_bit_cast(unsigned int, f);
  unsigned int r = u + 0x7FFFu + ((u >> 16) & 1u);
  return (unsigned short)(r >> 16);
}

// ---------------------------------------------------------------------------
// f32 -> bf16 bulk conversion (4 elems/thread, sizes always %4==0)
// ---------------------------------------------------------------------------
__global__ void cvt_bf16(const float* __restrict__ in, unsigned short* __restrict__ out,
                         size_t n4) {
  for (size_t i = blockIdx.x * (size_t)blockDim.x + threadIdx.x; i < n4;
       i += (size_t)gridDim.x * blockDim.x) {
    float4 v = reinterpret_cast<const float4*>(in)[i];
    ushort4 o;
    o.x = f2bf(v.x); o.y = f2bf(v.y); o.z = f2bf(v.z); o.w = f2bf(v.w);
    reinterpret_cast<ushort4*>(out)[i] = o;
  }
}

// W[Kd,Nn] f32 (row-major) -> Wp[Nn,Kd] bf16 (K contiguous per output column)
__global__ void pack_w(const float* __restrict__ W, unsigned short* __restrict__ Wp,
                       int Kd, int Nn) {
  size_t n = (size_t)Kd * Nn;
  for (size_t i = blockIdx.x * (size_t)blockDim.x + threadIdx.x; i < n;
       i += (size_t)gridDim.x * blockDim.x) {
    size_t col = i / Kd; int k = (int)(i - col * Kd);
    Wp[i] = f2bf(W[(size_t)k * Nn + col]);
  }
}

// ---------------------------------------------------------------------------
// GEMM: Y[M,Nn](f32) = A[M,Kd](bf16) @ Wp[Nn,Kd](bf16, pre-transposed).
// One wave computes a 16x64 tile: 1 A fragment reused by 4 B fragments.
// Requires M%16==0, Kd%32==0, Nn%64==0.  A/B fragments use contiguous k = 16*half + s.
// ---------------------------------------------------------------------------
__global__ void gemm_bf16v2(const unsigned short* __restrict__ A,
                            const unsigned short* __restrict__ Wp,
                            float* __restrict__ Y, int M, int Kd, int Nn) {
  const int lane   = threadIdx.x & 31;
  const int wave   = blockIdx.x * (blockDim.x >> 5) + (threadIdx.x >> 5);
  const int tilesN = Nn >> 6;                    // 64-wide N tiles
  const int tilesM = M >> 4;
  if (wave >= tilesM * tilesN) return;           // wave-uniform: EXEC stays all-ones
  const int tm   = wave / tilesN;
  const int tn   = wave - tm * tilesN;
  const int half = lane >> 4;                    // 0 or 1
  const int l15  = lane & 15;

  // A fragment: lane covers row tm*16+l15; per 32-K step needs
  // K = [half*16 .. half*16+7] and [half*16+8 .. half*16+15]  (two 16B loads).
  const unsigned short* __restrict__ arow = A + (size_t)(tm * 16 + l15) * Kd;
  // B fragments j=0..3: column n = tn*64 + j*16 + l15; K = half*16 .. half*16+15 (one 32B run).
  const unsigned short* __restrict__ b0 = Wp + (size_t)(tn * 64 +  0 + l15) * Kd;
  const unsigned short* __restrict__ b1 = Wp + (size_t)(tn * 64 + 16 + l15) * Kd;
  const unsigned short* __restrict__ b2 = Wp + (size_t)(tn * 64 + 32 + l15) * Kd;
  const unsigned short* __restrict__ b3 = Wp + (size_t)(tn * 64 + 48 + l15) * Kd;

  v8f acc0 = {}, acc1 = {}, acc2 = {}, acc3 = {};
  for (int k0 = 0; k0 < Kd; k0 += 32) {
    const v8u a0 = *reinterpret_cast<const v8u*>(arow + k0 + half * 16);
    const v8u a1 = *reinterpret_cast<const v8u*>(arow + k0 + half * 16 + 8);
    const v16u au = __builtin_shufflevector(a0, a1, 0, 1, 2, 3, 4, 5, 6, 7,
                                            8, 9, 10, 11, 12, 13, 14, 15);
    const v16bf a = __builtin_bit_cast(v16bf, au);
    const int kb = k0 + half * 16;
    const v16bf bb0 = __builtin_bit_cast(v16bf, *reinterpret_cast<const v16u*>(b0 + kb));
    const v16bf bb1 = __builtin_bit_cast(v16bf, *reinterpret_cast<const v16u*>(b1 + kb));
    const v16bf bb2 = __builtin_bit_cast(v16bf, *reinterpret_cast<const v16u*>(b2 + kb));
    const v16bf bb3 = __builtin_bit_cast(v16bf, *reinterpret_cast<const v16u*>(b3 + kb));
    acc0 = __builtin_amdgcn_wmma_f32_16x16x32_bf16(false, a, false, bb0, (short)0, acc0, false, false);
    acc1 = __builtin_amdgcn_wmma_f32_16x16x32_bf16(false, a, false, bb1, (short)0, acc1, false, false);
    acc2 = __builtin_amdgcn_wmma_f32_16x16x32_bf16(false, a, false, bb2, (short)0, acc2, false, false);
    acc3 = __builtin_amdgcn_wmma_f32_16x16x32_bf16(false, a, false, bb3, (short)0, acc3, false, false);
  }
  // C/D layout: lanes0-15 N=lane, VGPR v -> M=v ; lanes16-31 -> M=8+v.
  float* yrow = Y + (size_t)(tm * 16 + half * 8) * Nn + tn * 64 + l15;
#pragma unroll
  for (int v = 0; v < 8; ++v) {
    yrow[(size_t)v * Nn +  0] = acc0[v];
    yrow[(size_t)v * Nn + 16] = acc1[v];
    yrow[(size_t)v * Nn + 32] = acc2[v];
    yrow[(size_t)v * Nn + 48] = acc3[v];
  }
}

// ---------------------------------------------------------------------------
// Column stats (sum, sumsq) for training-mode BatchNorm.  blockDim.x == Nn.
// ---------------------------------------------------------------------------
__global__ void colstats(const float* __restrict__ Y, float* __restrict__ sum,
                         float* __restrict__ sumsq, int M, int Nn, int rowsPerBlock) {
  const int col = threadIdx.x;
  if (col >= Nn) return;
  int r0 = blockIdx.x * rowsPerBlock;
  int r1 = r0 + rowsPerBlock; if (r1 > M) r1 = M;
  float s = 0.f, q = 0.f;
  for (int r = r0; r < r1; ++r) {
    float v = Y[(size_t)r * Nn + col];
    s += v; q += v * v;
  }
  atomicAdd(&sum[col], s);
  atomicAdd(&sumsq[col], q);
}

__global__ void bn_apply(float* __restrict__ Y, const float* __restrict__ sum,
                         const float* __restrict__ sumsq, const float* __restrict__ g,
                         const float* __restrict__ b, int M, int Nn, size_t n) {
  for (size_t i = blockIdx.x * (size_t)blockDim.x + threadIdx.x; i < n;
       i += (size_t)gridDim.x * blockDim.x) {
    int col = (int)(i % Nn);
    float mean = sum[col] / (float)M;
    float var  = sumsq[col] / (float)M - mean * mean;
    if (var < 0.f) var = 0.f;
    float y = (Y[i] - mean) * rsqrtf(var + BN_EPS) * g[col] + b[col];
    Y[i] = y > 0.f ? y : 0.f;
  }
}

// ---------------------------------------------------------------------------
// Data-movement / elementwise kernels
// ---------------------------------------------------------------------------
__global__ void copy_into(float* __restrict__ dst, int dstW, int colOff,
                          const float* __restrict__ src, int srcW, int rows, int w) {
  size_t n = (size_t)rows * w;
  for (size_t i = blockIdx.x * (size_t)blockDim.x + threadIdx.x; i < n;
       i += (size_t)gridDim.x * blockDim.x) {
    size_t r = i / w; int c = (int)(i - r * w);
    dst[r * dstW + colOff + c] = src[r * srcW + c];
  }
}

// dst[r, colOff+c] = src[idx[r], c] (/ cnt[idx[r]] if cnt)
__global__ void gather_into(float* __restrict__ dst, int dstW, int colOff,
                            const float* __restrict__ src, const int* __restrict__ idx,
                            const float* __restrict__ cnt, int rows, int w) {
  size_t n = (size_t)rows * w;
  for (size_t i = blockIdx.x * (size_t)blockDim.x + threadIdx.x; i < n;
       i += (size_t)gridDim.x * blockDim.x) {
    size_t r = i / w; int c = (int)(i - r * w);
    int s = idx[r];
    float v = src[(size_t)s * w + c];
    if (cnt) v /= cnt[s];
    dst[r * dstW + colOff + c] = v;
  }
}

// cyc embedding: out[r] = emb[x[cyc_node[r]]]
__global__ void embed2(float* __restrict__ out, const float* __restrict__ emb,
                       const int* __restrict__ x, const int* __restrict__ cn,
                       int rows, int w) {
  size_t n = (size_t)rows * w;
  for (size_t i = blockIdx.x * (size_t)blockDim.x + threadIdx.x; i < n;
       i += (size_t)gridDim.x * blockDim.x) {
    size_t r = i / w; int c = (int)(i - r * w);
    out[i] = emb[(size_t)x[cn[r]] * w + c];
  }
}

// dst[sidx[r]] += src[gidx ? gidx[r] : r], rowwise
__global__ void gather_scatter_add(const float* __restrict__ src, const int* __restrict__ gidx,
                                   const int* __restrict__ sidx, float* __restrict__ dst,
                                   int rows, int w) {
  size_t n = (size_t)rows * w;
  for (size_t i = blockIdx.x * (size_t)blockDim.x + threadIdx.x; i < n;
       i += (size_t)gridDim.x * blockDim.x) {
    size_t r = i / w; int c = (int)(i - r * w);
    int gr = gidx ? gidx[r] : (int)r;
    atomicAdd(&dst[(size_t)sidx[r] * w + c], src[(size_t)gr * w + c]);
  }
}

// fixed-size sorted segments: out[s][c] = sum_{j<K} in[s*K+j][c]  (cyc_cid == r/K)
__global__ void seg_fixedK(const float* __restrict__ in, float* __restrict__ out,
                           int segs, int k, int w) {
  size_t n = (size_t)segs * w;
  for (size_t i = blockIdx.x * (size_t)blockDim.x + threadIdx.x; i < n;
       i += (size_t)gridDim.x * blockDim.x) {
    size_t s = i / w; int c = (int)(i - s * w);
    float acc = 0.f;
    const float* p = in + (s * k) * (size_t)w + c;
    for (int j = 0; j < k; ++j) acc += p[(size_t)j * w];
    out[i] = acc;
  }
}

__global__ void count_seg(const int* __restrict__ cid, float* __restrict__ cnt, int rows) {
  for (size_t i = blockIdx.x * (size_t)blockDim.x + threadIdx.x; i < (size_t)rows;
       i += (size_t)gridDim.x * blockDim.x)
    atomicAdd(&cnt[cid[i]], 1.f);
}

// out = (1+*e)*x + y
__global__ void combine2(float* __restrict__ out, const float* __restrict__ x,
                         const float* __restrict__ e, const float* __restrict__ y, size_t n) {
  float a = 1.f + e[0];
  for (size_t i = blockIdx.x * (size_t)blockDim.x + threadIdx.x; i < n;
       i += (size_t)gridDim.x * blockDim.x)
    out[i] = a * x[i] + y[i];
}

// out = (1+*e1)*x + (1+*e2)*y + z
__global__ void combine3(float* __restrict__ out, const float* __restrict__ x,
                         const float* __restrict__ e1, const float* __restrict__ y,
                         const float* __restrict__ e2, const float* __restrict__ z, size_t n) {
  float a = 1.f + e1[0], b = 1.f + e2[0];
  for (size_t i = blockIdx.x * (size_t)blockDim.x + threadIdx.x; i < n;
       i += (size_t)gridDim.x * blockDim.x)
    out[i] = a * x[i] + b * y[i] + z[i];
}

// out[R,2H] = (1+*e)*[cyc | (seg/cnt)[cid]] + lift1
__global__ void cyc_pre(float* __restrict__ out, const float* __restrict__ cyc,
                        const float* __restrict__ seg, const int* __restrict__ cid,
                        const float* __restrict__ cnt, const float* __restrict__ lift1,
                        const float* __restrict__ e, int rows) {
  float a = 1.f + e[0];
  size_t n = (size_t)rows * H2;
  for (size_t i = blockIdx.x * (size_t)blockDim.x + threadIdx.x; i < n;
       i += (size_t)gridDim.x * blockDim.x) {
    size_t r = i / H2; int c = (int)(i - r * H2);
    float lm;
    if (c < HH) lm = cyc[r * HH + c];
    else { int s = cid[r]; lm = seg[(size_t)s * HH + (c - HH)] / cnt[s]; }
    out[i] = a * lm + lift1[i];
  }
}

__global__ void add3(float* __restrict__ out, const float* __restrict__ a,
                     const float* __restrict__ b, const float* __restrict__ c, size_t n) {
  for (size_t i = blockIdx.x * (size_t)blockDim.x + threadIdx.x; i < n;
       i += (size_t)gridDim.x * blockDim.x)
    out[i] = a[i] + b[i] + c[i];
}

// out[g] = dot(rep[g,:], W[:,0]) + b
__global__ void head(const float* __restrict__ rep, const float* __restrict__ W,
                     const float* __restrict__ b, float* __restrict__ out, int g_, int k2) {
  int g = blockIdx.x * blockDim.x + threadIdx.x;
  if (g >= g_) return;
  float s = b[0];
  const float* r = rep + (size_t)g * k2;
  for (int k = 0; k < k2; ++k) s += r[k] * W[k];
  out[g] = s;
}

// ---------------------------------------------------------------------------
// Host-side helpers
// ---------------------------------------------------------------------------
static inline int gridFor(size_t n, int thr) {
  size_t b = (n + thr - 1) / thr;
  if (b > (size_t)1 << 28) b = (size_t)1 << 28;
  return (int)b;
}

// Full GEMM: convert A to bf16, pack W transposed bf16, run WMMA GEMM.
static void gemm_run(const float* A, const float* W, float* Y, int M, int Kd, int Nn,
                     unsigned short* ABF, unsigned short* WP, hipStream_t s) {
  size_t n4 = (size_t)M * Kd / 4;
  cvt_bf16<<<gridFor(n4, 256), 256, 0, s>>>(A, ABF, n4);
  pack_w<<<gridFor((size_t)Nn * Kd, 256), 256, 0, s>>>(W, WP, Kd, Nn);
  int tiles = (M >> 4) * (Nn >> 6);
  gemm_bf16v2<<<(tiles + 7) / 8, 256, 0, s>>>(ABF, WP, Y, M, Kd, Nn);
}

static void bnrelu(float* Y, const float* g, const float* b, int M, int Nn, float* stat,
                   hipStream_t s) {
  hipMemsetAsync(stat, 0, (size_t)2 * Nn * sizeof(float), s);
  int rpb = 1024;
  colstats<<<(M + rpb - 1) / rpb, Nn, 0, s>>>(Y, stat, stat + Nn, M, Nn, rpb);
  size_t n = (size_t)M * Nn;
  bn_apply<<<gridFor(n, 256), 256, 0, s>>>(Y, stat, stat + Nn, g, b, M, Nn, n);
}

// ---------------------------------------------------------------------------
// Orchestration.  Param flattening (JAX pytree, dict keys sorted):
//  0..9 data arrays; 10 cyc_emb; 11 edge_emb;
//  per layer (38): ec_eps11,ec_eps12,ec_eps2, ec_lift{W1,W2,b1,b2,g1,g2},
//   ec_lvl1{W,b,g}, ec_lvl2{W1,W2,b1,b2,g1,g2}, fuse{W,b,g}, ne_eps1,ne_eps2,
//   ne_lift{6}, ne_lvl1{3}, ne_lvl2{6};
//  then lin_W, lin_b, node_emb, pool[0..2]{W,b,g}.
// ---------------------------------------------------------------------------
extern "C" void kernel_launch(void* const* d_in, const int* in_sizes, int n_in,
                              void* d_out, int out_size, void* d_ws, size_t ws_size,
                              hipStream_t stream) {
  auto F = [&](int i) { return (const float*)d_in[i]; };
  auto I = [&](int i) { return (const int*)d_in[i]; };

  const int* x          = I(0);
  const int* edge_attr  = I(1);
  const int* inc_node   = I(2);
  const int* inc_edge   = I(3);
  const int* cyc_edge   = I(4);
  const int* cyc_cid    = I(5);
  const int* cyc_node   = I(6);
  const int* node2graph = I(7);
  const int* edge2graph = I(8);
  const int* cyc2graph  = I(9);
  const float* cyc_emb  = F(10);
  const float* edge_emb = F(11);
  const float* lin_W    = F(126);
  const float* lin_b    = F(127);
  const float* node_emb = F(128);

  // ---- workspace carve-up (floats) ----
  float* ws = (float*)d_ws;
  size_t off = 0;
  auto alloc = [&](size_t n) { float* p = ws + off; off += n; return p; };
  float* node  = alloc((size_t)NN * HH);
  float* edge  = alloc((size_t)EE * HH);
  float* cyc   = alloc((size_t)RR * HH);
  float* P0    = alloc((size_t)EE * HH);   // generic width-H buffers (rows <= E)
  float* P1    = alloc((size_t)EE * HH);
  float* P2    = alloc((size_t)EE * HH);
  float* LIFT1 = alloc((size_t)RR * H2);
  float* BIG0  = alloc((size_t)EE * H2);   // also covers R*3H
  float* BIG1  = alloc((size_t)EE * H2);
  float* SEG0  = alloc((size_t)CC * HH);
  float* CNT   = alloc(CC);
  float* STAT  = alloc(512);
  float* PN    = alloc((size_t)GG * HH);
  float* PE    = alloc((size_t)GG * HH);
  float* PC    = alloc((size_t)GG * HH);
  float* POOLA = alloc((size_t)GG * H2);
  float* POOLB = alloc((size_t)GG * H2);
  float* POOLC = alloc((size_t)GG * H2);
  float* REP   = alloc((size_t)GG * H2);
  unsigned short* ABF = (unsigned short*)alloc((size_t)EE * H2 / 2); // bf16 A operand
  unsigned short* WP  = (unsigned short*)alloc(((size_t)H3 * H2) / 2 + 64); // bf16 packed W
  if (off * sizeof(float) > ws_size) return;  // workspace too small

  const int T = 256;
  hipStream_t s = stream;

  // ---- embeddings ----
  gather_into<<<gridFor((size_t)NN * HH, T), T, 0, s>>>(node, HH, 0, node_emb, x, nullptr, NN, HH);
  gather_into<<<gridFor((size_t)EE * HH, T), T, 0, s>>>(edge, HH, 0, edge_emb, edge_attr, nullptr, EE, HH);
  embed2<<<gridFor((size_t)RR * HH, T), T, 0, s>>>(cyc, cyc_emb, x, cyc_node, RR, HH);
  hipMemsetAsync(CNT, 0, CC * sizeof(float), s);
  count_seg<<<gridFor(RR, T), T, 0, s>>>(cyc_cid, CNT, RR);

  for (int l = 0; l < 3; ++l) {
    const int B = 12 + 38 * l;
    const float *ec_e11 = F(B + 0), *ec_e12 = F(B + 1), *ec_e2 = F(B + 2);
    const float *eclift_W1 = F(B + 3), *eclift_W2 = F(B + 4), *eclift_b1 = F(B + 5),
                *eclift_b2 = F(B + 6), *eclift_g1 = F(B + 7), *eclift_g2 = F(B + 8);
    const float *eclvl1_W = F(B + 9), *eclvl1_b = F(B + 10), *eclvl1_g = F(B + 11);
    const float *eclvl2_W1 = F(B + 12), *eclvl2_W2 = F(B + 13), *eclvl2_b1 = F(B + 14),
                *eclvl2_b2 = F(B + 15), *eclvl2_g1 = F(B + 16), *eclvl2_g2 = F(B + 17);
    const float *fuse_W = F(B + 18), *fuse_b = F(B + 19), *fuse_g = F(B + 20);
    const float *ne_e1 = F(B + 21), *ne_e2 = F(B + 22);
    const float *nelift_W1 = F(B + 23), *nelift_W2 = F(B + 24), *nelift_b1 = F(B + 25),
                *nelift_b2 = F(B + 26), *nelift_g1 = F(B + 27), *nelift_g2 = F(B + 28);
    const float *nelvl1_W = F(B + 29), *nelvl1_b = F(B + 30), *nelvl1_g = F(B + 31);
    const float *nelvl2_W1 = F(B + 32), *nelvl2_W2 = F(B + 33), *nelvl2_b1 = F(B + 34),
                *nelvl2_b2 = F(B + 35), *nelvl2_g1 = F(B + 36), *nelvl2_g2 = F(B + 37);

    // lift = seg(node[inc_node], inc_edge, E)  -> P0
    hipMemsetAsync(P0, 0, (size_t)EE * HH * sizeof(float), s);
    gather_scatter_add<<<gridFor((size_t)2 * EE * HH, T), T, 0, s>>>(node, inc_node, inc_edge, P0, 2 * EE, HH);
    // lvl_e = mlp1([lift|edge]) -> P1
    copy_into<<<gridFor((size_t)EE * HH, T), T, 0, s>>>(BIG0, H2, 0, P0, HH, EE, HH);
    copy_into<<<gridFor((size_t)EE * HH, T), T, 0, s>>>(BIG0, H2, HH, edge, HH, EE, HH);
    gemm_run(BIG0, nelvl1_W, P1, EE, H2, HH, ABF, WP, s);
    bnrelu(P1, nelvl1_g, nelvl1_b, EE, HH, STAT, s);
    // lvl_n = seg(lvl_e[inc_edge], inc_node, N) -> P2
    hipMemsetAsync(P2, 0, (size_t)NN * HH * sizeof(float), s);
    gather_scatter_add<<<gridFor((size_t)2 * EE * HH, T), T, 0, s>>>(P1, inc_edge, inc_node, P2, 2 * EE, HH);
    // node_out = mlp2((1+e1)*node + lvl_n) -> P2
    combine2<<<gridFor((size_t)NN * HH, T), T, 0, s>>>(P2, node, ne_e1, P2, (size_t)NN * HH);
    gemm_run(P2, nelvl2_W1, BIG1, NN, HH, H2, ABF, WP, s);
    bnrelu(BIG1, nelvl2_g1, nelvl2_b1, NN, H2, STAT, s);
    gemm_run(BIG1, nelvl2_W2, P2, NN, H2, HH, ABF, WP, s);
    bnrelu(P2, nelvl2_g2, nelvl2_b2, NN, HH, STAT, s);
    // edge_out1 = mlp2((1+e2)*edge + lift) -> P0
    combine2<<<gridFor((size_t)EE * HH, T), T, 0, s>>>(P0, edge, ne_e2, P0, (size_t)EE * HH);
    gemm_run(P0, nelift_W1, BIG1, EE, HH, H2, ABF, WP, s);
    bnrelu(BIG1, nelift_g1, nelift_b1, EE, H2, STAT, s);
    gemm_run(BIG1, nelift_W2, P0, EE, H2, HH, ABF, WP, s);
    bnrelu(P0, nelift_g2, nelift_b2, EE, HH, STAT, s);

    // msg = edge[cyc_edge] -> P1 ; segmsg -> SEG0 ; lift1 = [msg | segmsg[cid]] -> LIFT1
    gather_into<<<gridFor((size_t)RR * HH, T), T, 0, s>>>(P1, HH, 0, edge, cyc_edge, nullptr, RR, HH);
    seg_fixedK<<<gridFor((size_t)CC * HH, T), T, 0, s>>>(P1, SEG0, CC, KK, HH);
    copy_into<<<gridFor((size_t)RR * HH, T), T, 0, s>>>(LIFT1, H2, 0, P1, HH, RR, HH);
    gather_into<<<gridFor((size_t)RR * HH, T), T, 0, s>>>(LIFT1, H2, HH, SEG0, cyc_cid, nullptr, RR, HH);
    // lvl_c = mlp1([lift1|cyc]) -> P1
    copy_into<<<gridFor((size_t)RR * H2, T), T, 0, s>>>(BIG0, H3, 0, LIFT1, H2, RR, H2);
    copy_into<<<gridFor((size_t)RR * HH, T), T, 0, s>>>(BIG0, H3, H2, cyc, HH, RR, HH);
    gemm_run(BIG0, eclvl1_W, P1, RR, H3, HH, ABF, WP, s);
    bnrelu(P1, eclvl1_g, eclvl1_b, RR, HH, STAT, s);
    // back0 -> BIG1[0:E*H], back1 -> BIG1[E*H:]
    float* back0 = BIG1;
    float* back1 = BIG1 + (size_t)EE * HH;
    seg_fixedK<<<gridFor((size_t)CC * HH, T), T, 0, s>>>(P1, SEG0, CC, KK, HH);
    hipMemsetAsync(back0, 0, (size_t)EE * HH * sizeof(float), s);
    gather_scatter_add<<<gridFor((size_t)RR * HH, T), T, 0, s>>>(P1, nullptr, cyc_edge, back0, RR, HH);
    gather_into<<<gridFor((size_t)RR * HH, T), T, 0, s>>>(P1, HH, 0, SEG0, cyc_cid, CNT, RR, HH);
    hipMemsetAsync(back1, 0, (size_t)EE * HH * sizeof(float), s);
    gather_scatter_add<<<gridFor((size_t)RR * HH, T), T, 0, s>>>(P1, nullptr, cyc_edge, back1, RR, HH);
    // edge_out2 = mlp2((1+e11)*edge + (1+e12)*back0 + back1) -> BIG0[0:E*H]
    combine3<<<gridFor((size_t)EE * HH, T), T, 0, s>>>(BIG0, edge, ec_e11, back0, ec_e12, back1, (size_t)EE * HH);
    gemm_run(BIG0, eclvl2_W1, BIG1, EE, HH, H2, ABF, WP, s);
    bnrelu(BIG1, eclvl2_g1, eclvl2_b1, EE, H2, STAT, s);
    gemm_run(BIG1, eclvl2_W2, BIG0, EE, H2, HH, ABF, WP, s);
    bnrelu(BIG0, eclvl2_g2, eclvl2_b2, EE, HH, STAT, s);
    // cyc = mlp2((1+e2)*[cyc|mean] + lift1)
    seg_fixedK<<<gridFor((size_t)CC * HH, T), T, 0, s>>>(cyc, SEG0, CC, KK, HH);
    cyc_pre<<<gridFor((size_t)RR * H2, T), T, 0, s>>>(BIG1, cyc, SEG0, cyc_cid, CNT, LIFT1, ec_e2, RR);
    gemm_run(BIG1, eclift_W1, LIFT1, RR, H2, H2, ABF, WP, s);
    bnrelu(LIFT1, eclift_g1, eclift_b1, RR, H2, STAT, s);
    gemm_run(LIFT1, eclift_W2, P1, RR, H2, HH, ABF, WP, s);
    bnrelu(P1, eclift_g2, eclift_b2, RR, HH, STAT, s);            // cyc_new = P1
    // edge = mlp1([edge_out1 | edge_out2]) -> P0
    copy_into<<<gridFor((size_t)EE * HH, T), T, 0, s>>>(BIG1, H2, 0, P0, HH, EE, HH);
    copy_into<<<gridFor((size_t)EE * HH, T), T, 0, s>>>(BIG1, H2, HH, BIG0, HH, EE, HH);
    gemm_run(BIG1, fuse_W, P0, EE, H2, HH, ABF, WP, s);
    bnrelu(P0, fuse_g, fuse_b, EE, HH, STAT, s);
    // commit
    hipMemcpyAsync(node, P2, (size_t)NN * HH * sizeof(float), hipMemcpyDeviceToDevice, s);
    hipMemcpyAsync(edge, P0, (size_t)EE * HH * sizeof(float), hipMemcpyDeviceToDevice, s);
    hipMemcpyAsync(cyc,  P1, (size_t)RR * HH * sizeof(float), hipMemcpyDeviceToDevice, s);
  }

  // ---- readout ----
  hipMemsetAsync(PN, 0, (size_t)GG * HH * sizeof(float), s);
  hipMemsetAsync(PE, 0, (size_t)GG * HH * sizeof(float), s);
  hipMemsetAsync(PC, 0, (size_t)GG * HH * sizeof(float), s);
  gather_scatter_add<<<gridFor((size_t)NN * HH, T), T, 0, s>>>(node, nullptr, node2graph, PN, NN, HH);
  gather_scatter_add<<<gridFor((size_t)EE * HH, T), T, 0, s>>>(edge, nullptr, edge2graph, PE, EE, HH);
  gather_scatter_add<<<gridFor((size_t)RR * HH, T), T, 0, s>>>(cyc, nullptr, cyc2graph, PC, RR, HH);

  const float *p0W = F(129), *p0b = F(130), *p0g = F(131);
  const float *p1W = F(132), *p1b = F(133), *p1g = F(134);
  const float *p2W = F(135), *p2b = F(136), *p2g = F(137);
  gemm_run(PN, p0W, POOLA, GG, HH, H2, ABF, WP, s); bnrelu(POOLA, p0g, p0b, GG, H2, STAT, s);
  gemm_run(PE, p1W, POOLB, GG, HH, H2, ABF, WP, s); bnrelu(POOLB, p1g, p1b, GG, H2, STAT, s);
  gemm_run(PC, p2W, POOLC, GG, HH, H2, ABF, WP, s); bnrelu(POOLC, p2g, p2b, GG, H2, STAT, s);
  add3<<<gridFor((size_t)GG * H2, T), T, 0, s>>>(REP, POOLA, POOLB, POOLC, (size_t)GG * H2);
  head<<<(GG + T - 1) / T, T, 0, s>>>(REP, lin_W, lin_b, (float*)d_out, GG, H2);
}
